// VAECW_40072044871848
// MI455X (gfx1250) — matrix operand order repaired and checked by reading, not compiled, not run on hardware
//
#include <hip/hip_runtime.h>
#include <hip/hip_bf16.h>

// ---------------------------------------------------------------------------
// VAE codeword model on MI455X (gfx1250): all matmuls via v_wmma_f32_16x16x32_bf16
// 4 N-tiles per wave (A-fragment reuse, 4 independent accumulators).
// ---------------------------------------------------------------------------

typedef __attribute__((ext_vector_type(16))) __bf16 v16bf;
typedef __attribute__((ext_vector_type(8)))  float  v8f;
typedef __attribute__((ext_vector_type(4)))  float  v4f;

#define SLOPE 0.2f

__device__ __forceinline__ v16bf load_a_frag(const float* __restrict__ p) {
    // p -> A[row] + k0 + 16*half ; elements 0..15 = p[0..15]
    v16bf a;
#pragma unroll
    for (int j = 0; j < 8; ++j) a[j] = (__bf16)p[j];
#pragma unroll
    for (int j = 0; j < 8; ++j) a[j + 8] = (__bf16)p[8 + j];
    return a;
}

__device__ __forceinline__ v16bf load_b_frag(const __bf16* __restrict__ p) {
    // p -> W[col] + k0 + 16*half ; 16 contiguous bf16 (32B aligned)
    return *(const v16bf*)p;
}

__device__ __forceinline__ v8f wmma_bf16(v16bf a, v16bf b, v8f c) {
    return __builtin_amdgcn_wmma_f32_16x16x32_bf16(false, a, false, b, (short)0, c,
                                                   false, false);
}

// ------------------- generic GEMM: C = act(A*W^T + bias) -------------------
// One wave32 per 16x64 C strip (4 N-tiles). A: fp32 (M x K) row-major.
// W: bf16 (N x K) row-major. N must be a multiple of 64.
template <int ACT>
__global__ void wmma_gemm4(const float* __restrict__ A, int lda,
                           const __bf16* __restrict__ W, int ldw,
                           const float* __restrict__ bias,
                           float* __restrict__ C, int ldc,
                           int Mtiles, int Ngroups, int K) {
    int wave = (blockIdx.x * blockDim.x + threadIdx.x) >> 5;
    if (wave >= Mtiles * Ngroups) return;         // wave-uniform: EXEC stays all-1
    int lane = threadIdx.x & 31;
    int mt = wave % Mtiles, ng = wave / Mtiles;
    int half = lane >> 4, lr = lane & 15;
    int row = mt * 16 + lr;
    int col0 = ng * 64 + lr;

    v8f acc[4] = {{}, {}, {}, {}};
    const float* arow = A + (size_t)row * lda;
    for (int k0 = 0; k0 < K; k0 += 32) {
        if (k0 + 32 < K) __builtin_prefetch(arow + k0 + 32 + 8 * half, 0, 1);
        v16bf a = load_a_frag(arow + k0 + 16 * half);
#pragma unroll
        for (int t = 0; t < 4; ++t) {
            v16bf b = load_b_frag(W + (size_t)(col0 + t * 16) * ldw + k0 + 16 * half);
            acc[t] = wmma_bf16(a, b, acc[t]);
        }
    }
#pragma unroll
    for (int t = 0; t < 4; ++t) {
        int col = col0 + t * 16;
        float bn = bias ? bias[col] : 0.0f;
#pragma unroll
        for (int r = 0; r < 8; ++r) {
            float v = acc[t][r] + bn;
            if (ACT) v = v > 0.0f ? v : SLOPE * v;
            C[(size_t)(mt * 16 + r + 8 * half) * ldc + col] = v;
        }
    }
}

// ----------------- encoder: lrelu(xe @ enc_w1^T + b1), max over 64 codes ----
// xe: (B*64, 64) view of x. One wave per (b, 16-col tile); B-frags hoisted.
__global__ void enc_fused(const float* __restrict__ x,
                          const __bf16* __restrict__ W,   // (512,64) bf16
                          const float* __restrict__ bias, // (512)
                          float* __restrict__ out) {      // (512,512)
    int wave = (blockIdx.x * blockDim.x + threadIdx.x) >> 5;  // 0..16383
    int lane = threadIdx.x & 31;
    int b = wave >> 5;
    int nt = wave & 31;
    int half = lane >> 4, lr = lane & 15;
    int col = nt * 16 + lr;
    float bn = bias[col];
    float vmax = -__builtin_inff();

    v16bf w0 = load_b_frag(W + (size_t)col * 64 + 16 * half);        // K 0..31
    v16bf w1 = load_b_frag(W + (size_t)col * 64 + 32 + 16 * half);   // K 32..63

#pragma unroll
    for (int ct = 0; ct < 4; ++ct) {
        int row = b * 64 + ct * 16 + lr;
        v8f acc = {};
        acc = wmma_bf16(load_a_frag(x + (size_t)row * 64 + 16 * half), w0, acc);
        acc = wmma_bf16(load_a_frag(x + (size_t)row * 64 + 32 + 16 * half), w1, acc);
#pragma unroll
        for (int r = 0; r < 8; ++r) {
            float v = acc[r] + bn;
            v = v > 0.0f ? v : SLOPE * v;
            vmax = fmaxf(vmax, v);
        }
    }
    vmax = fmaxf(vmax, __shfl_xor(vmax, 16, 32));   // combine row-halves
    if (half == 0) out[(size_t)b * 512 + col] = vmax;
}

// -------- dist: d[b][c][s] = |xr|^2 + |book|^2 - 2 * <xr[b,c,:], book[c,s,:]> -
// One wave per (c, 16-row, 64-col strip): 4 N-tiles, A-frag reused.
__global__ void dist_wmma4(const float* __restrict__ recon,   // (512,4096)
                           const __bf16* __restrict__ bookb,  // (64,1024,64)
                           const float* __restrict__ rnorm,   // (512,64)
                           const float* __restrict__ bnorm,   // (64,1024)
                           float* __restrict__ dist) {        // (512,64,1024)
    int wave = (blockIdx.x * blockDim.x + threadIdx.x) >> 5;  // 0..32767
    int lane = threadIdx.x & 31;
    int c = wave >> 9;          // /(32*16)
    int rem = wave & 511;
    int ntg = rem >> 5;         // 0..15
    int mt = rem & 31;          // 0..31
    int half = lane >> 4, lr = lane & 15;
    int row = mt * 16 + lr;     // b
    int col0 = ntg * 64 + lr;   // s base

    v8f acc[4] = {{}, {}, {}, {}};
#pragma unroll
    for (int k0 = 0; k0 < 64; k0 += 32) {
        v16bf a = load_a_frag(recon + (size_t)row * 4096 + c * 64 + k0 + 16 * half);
#pragma unroll
        for (int t = 0; t < 4; ++t) {
            v16bf w = load_b_frag(bookb + ((size_t)c * 1024 + col0 + t * 16) * 64 +
                                  k0 + 16 * half);
            acc[t] = wmma_bf16(a, w, acc[t]);
        }
    }
    float rn[8];
#pragma unroll
    for (int r = 0; r < 8; ++r) rn[r] = rnorm[(mt * 16 + r + 8 * half) * 64 + c];
#pragma unroll
    for (int t = 0; t < 4; ++t) {
        int col = col0 + t * 16;
        float bn = bnorm[c * 1024 + col];
#pragma unroll
        for (int r = 0; r < 8; ++r) {
            int b = mt * 16 + r + 8 * half;
            dist[((size_t)b * 64 + c) * 1024 + col] = rn[r] + bn - 2.0f * acc[t][r];
        }
    }
}

// ------------------------------- helpers ----------------------------------
__global__ void cvt_bf16(const float* __restrict__ src, __bf16* __restrict__ dst, int n) {
    int i = blockIdx.x * blockDim.x + threadIdx.x;
    if (i < n) dst[i] = (__bf16)src[i];
}

__global__ void rowsumsq64(const float* __restrict__ src, float* __restrict__ dst, int nrows) {
    int i = blockIdx.x * blockDim.x + threadIdx.x;
    if (i >= nrows) return;
    const v4f* p = (const v4f*)(src + (size_t)i * 64);
    float s = 0.0f;
#pragma unroll
    for (int j = 0; j < 16; ++j) {
        v4f q = p[j];
        s += q[0] * q[0] + q[1] * q[1] + q[2] * q[2] + q[3] * q[3];
    }
    dst[i] = s;
}

__global__ void split_half(const float* __restrict__ t, float* __restrict__ lo,
                           float* __restrict__ hi, int rows, int halfc) {
    int i = blockIdx.x * blockDim.x + threadIdx.x;
    int total = rows * 2 * halfc;
    if (i >= total) return;
    int b = i / (2 * halfc);
    int col = i - b * 2 * halfc;
    float v = t[i];
    if (col < halfc) lo[(size_t)b * halfc + col] = v;
    else             hi[(size_t)b * halfc + col - halfc] = v;
}

__global__ void build_cat(const float* __restrict__ mu, const float* __restrict__ h,
                          float* __restrict__ cat) {
    int i = blockIdx.x * blockDim.x + threadIdx.x;
    if (i >= 512 * 2304) return;
    int b = i / 2304;
    int col = i - b * 2304;
    cat[i] = (col < 256) ? mu[(size_t)b * 256 + col] : h[(size_t)b * 2048 + (col - 256)];
}

__global__ void vadd(const float* __restrict__ a, const float* __restrict__ b,
                     float* __restrict__ c, int n) {
    int i = blockIdx.x * blockDim.x + threadIdx.x;
    if (i < n) c[i] = a[i] + b[i];
}

// one wave per (b,c) row of 1024; float4 strided scan + shfl reduction.
__global__ void argmin_rows(const float* __restrict__ dist, float* __restrict__ idx_out,
                            int nrows) {
    int wave = (blockIdx.x * blockDim.x + threadIdx.x) >> 5;
    int lane = threadIdx.x & 31;
    if (wave >= nrows) return;
    const float* p = dist + (size_t)wave * 1024;
    float best = __builtin_inff();
    int bi = 0;
#pragma unroll
    for (int it = 0; it < 8; ++it) {
        int s = it * 128 + lane * 4;
        v4f q = *(const v4f*)(p + s);
#pragma unroll
        for (int j = 0; j < 4; ++j) {
            if (q[j] < best) { best = q[j]; bi = s + j; }
        }
    }
#pragma unroll
    for (int d = 16; d >= 1; d >>= 1) {
        float ov = __shfl_xor(best, d, 32);
        int   oi = __shfl_xor(bi, d, 32);
        if (ov < best || (ov == best && oi < bi)) { best = ov; bi = oi; }
    }
    if (lane == 0) idx_out[wave] = (float)bi;
}

// ---------------------------------------------------------------------------
extern "C" void kernel_launch(void* const* d_in, const int* in_sizes, int n_in,
                              void* d_out, int out_size, void* d_ws, size_t ws_size,
                              hipStream_t stream) {
    (void)in_sizes; (void)n_in; (void)out_size; (void)ws_size;
    const float* x        = (const float*)d_in[0];
    const float* codebook = (const float*)d_in[1];
    const float* enc_w1   = (const float*)d_in[2];
    const float* enc_b1   = (const float*)d_in[3];
    const float* enc_w2   = (const float*)d_in[4];
    const float* enc_b2   = (const float*)d_in[5];
    const float* inf1_w   = (const float*)d_in[6];
    const float* inf1_b   = (const float*)d_in[7];
    const float* inf2_w1  = (const float*)d_in[8];
    const float* inf2_b1  = (const float*)d_in[9];
    const float* inf2_w2  = (const float*)d_in[10];
    const float* inf2_b2  = (const float*)d_in[11];
    const float* prior_w1 = (const float*)d_in[12];
    const float* prior_b1 = (const float*)d_in[13];
    const float* prior_w2 = (const float*)d_in[14];
    const float* prior_b2 = (const float*)d_in[15];
    const float* dec_w1   = (const float*)d_in[16];
    const float* dec_b1   = (const float*)d_in[17];
    const float* dec_w2   = (const float*)d_in[18];
    const float* dec_b2   = (const float*)d_in[19];

    float* out = (float*)d_out;
    float* o_recon   = out;             // (512,4096)
    float* o_dist    = out + 2097152;   // (512,64,1024)
    float* o_idx     = out + 35651584;  // (512*64)
    float* o_mu      = out + 35684352;  // (512,256)
    float* o_logvar  = out + 35815424;  // (512,256)
    float* o_dmu     = out + 35946496;  // (512,768)
    float* o_dlogvar = out + 36339712;  // (512,768)
    float* o_plogvar = out + 36732928;  // (512,768)

    // ---- workspace carve-out (256B aligned) ----
    char* ws = (char*)d_ws;
    size_t off = 0;
    auto carve = [&](size_t bytes) -> char* {
        char* p = ws + off;
        off = (off + bytes + 255) & ~(size_t)255;
        return p;
    };
    __bf16* wb_enc_w1   = (__bf16*)carve((size_t)512 * 64 * 2);
    __bf16* wb_enc_w2   = (__bf16*)carve((size_t)2048 * 512 * 2);
    __bf16* wb_inf1     = (__bf16*)carve((size_t)512 * 2048 * 2);
    __bf16* wb_inf2_w1  = (__bf16*)carve((size_t)2048 * 2304 * 2);
    __bf16* wb_inf2_w2  = (__bf16*)carve((size_t)1536 * 2048 * 2);
    __bf16* wb_prior_w1 = (__bf16*)carve((size_t)2048 * 256 * 2);
    __bf16* wb_prior_w2 = (__bf16*)carve((size_t)1536 * 2048 * 2);
    __bf16* wb_dec_w1   = (__bf16*)carve((size_t)2048 * 768 * 2);
    __bf16* wb_dec_w2   = (__bf16*)carve((size_t)4096 * 2048 * 2);
    __bf16* wb_book     = (__bf16*)carve((size_t)64 * 1024 * 64 * 2);
    float* h_pool = (float*)carve((size_t)512 * 512 * 4);
    float* h      = (float*)carve((size_t)512 * 2048 * 4);
    float* t1     = (float*)carve((size_t)512 * 512 * 4);
    float* cat    = (float*)carve((size_t)512 * 2304 * 4);
    float* p1     = (float*)carve((size_t)512 * 2048 * 4);
    float* pbuf   = (float*)carve((size_t)512 * 1536 * 4);
    float* pmu    = (float*)carve((size_t)512 * 768 * 4);
    float* d1     = (float*)carve((size_t)512 * 2048 * 4);
    float* dfull  = (float*)carve((size_t)512 * 1536 * 4);
    float* z2     = (float*)carve((size_t)512 * 768 * 4);
    float* g1     = (float*)carve((size_t)512 * 2048 * 4);
    float* rnorm  = (float*)carve((size_t)512 * 64 * 4);
    float* bnorm  = (float*)carve((size_t)64 * 1024 * 4);

    auto cvt = [&](const float* s, __bf16* d, int n) {
        cvt_bf16<<<(n + 255) / 256, 256, 0, stream>>>(s, d, n);
    };
    cvt(enc_w1, wb_enc_w1, 512 * 64);
    cvt(enc_w2, wb_enc_w2, 2048 * 512);
    cvt(inf1_w, wb_inf1, 512 * 2048);
    cvt(inf2_w1, wb_inf2_w1, 2048 * 2304);
    cvt(inf2_w2, wb_inf2_w2, 1536 * 2048);
    cvt(prior_w1, wb_prior_w1, 2048 * 256);
    cvt(prior_w2, wb_prior_w2, 1536 * 2048);
    cvt(dec_w1, wb_dec_w1, 2048 * 768);
    cvt(dec_w2, wb_dec_w2, 4096 * 2048);
    cvt(codebook, wb_book, 64 * 1024 * 64);

    // N must be a multiple of 64: Ngroups = N/64
    auto gemm = [&](int act, const float* A, int lda, const __bf16* W, int ldw,
                    const float* bias, float* C, int ldc, int Mt, int Ng, int K) {
        int blocks = (Mt * Ng + 3) / 4;  // 4 waves / block
        if (act)
            wmma_gemm4<1><<<blocks, 128, 0, stream>>>(A, lda, W, ldw, bias, C, ldc, Mt, Ng, K);
        else
            wmma_gemm4<0><<<blocks, 128, 0, stream>>>(A, lda, W, ldw, bias, C, ldc, Mt, Ng, K);
    };

    // encode: conv + lrelu + max-pool  ->  h_pool (512,512)
    enc_fused<<<4096, 128, 0, stream>>>(x, wb_enc_w1, enc_b1, h_pool);
    // h = h_pool @ enc_w2^T + b2   (512,2048)
    gemm(0, h_pool, 512, wb_enc_w2, 512, enc_b2, h, 2048, 32, 32, 512);
    // t1 = h @ inf1_w^T + b        (512,512) -> mu | log_var
    gemm(0, h, 2048, wb_inf1, 2048, inf1_b, t1, 512, 32, 8, 2048);
    split_half<<<(512 * 512 + 255) / 256, 256, 0, stream>>>(t1, o_mu, o_logvar, 512, 256);
    build_cat<<<(512 * 2304 + 255) / 256, 256, 0, stream>>>(o_mu, h, cat);
    // prior path: p = lrelu(z1 @ pw1^T + b1) @ pw2^T + b2   (512,1536)
    gemm(1, o_mu, 256, wb_prior_w1, 256, prior_b1, p1, 2048, 32, 32, 256);
    gemm(0, p1, 2048, wb_prior_w2, 2048, prior_b2, pbuf, 1536, 32, 24, 2048);
    split_half<<<(512 * 1536 + 255) / 256, 256, 0, stream>>>(pbuf, pmu, o_plogvar, 512, 768);
    // inf2 path: d = lrelu(cat @ iw1^T + b1) @ iw2^T + b2   (512,1536)
    gemm(1, cat, 2304, wb_inf2_w1, 2304, inf2_b1, d1, 2048, 32, 32, 2304);
    gemm(0, d1, 2048, wb_inf2_w2, 2048, inf2_b2, dfull, 1536, 32, 24, 2048);
    split_half<<<(512 * 1536 + 255) / 256, 256, 0, stream>>>(dfull, o_dmu, o_dlogvar, 512, 768);
    vadd<<<(512 * 768 + 255) / 256, 256, 0, stream>>>(o_dmu, pmu, z2, 512 * 768);
    // decoder
    gemm(1, z2, 768, wb_dec_w1, 768, dec_b1, g1, 2048, 32, 32, 768);
    gemm(0, g1, 2048, wb_dec_w2, 2048, dec_b2, o_recon, 4096, 32, 64, 2048);
    // distances + argmin
    rowsumsq64<<<(32768 + 255) / 256, 256, 0, stream>>>(o_recon, rnorm, 32768);
    rowsumsq64<<<(65536 + 255) / 256, 256, 0, stream>>>(codebook, bnorm, 65536);
    dist_wmma4<<<8192, 128, 0, stream>>>(o_recon, wb_book, rnorm, bnorm, o_dist);
    argmin_rows<<<(32768 * 32 + 255) / 256, 256, 0, stream>>>(o_dist, o_idx, 32768);
}
